// MoDRouter_26998164423421
// MI455X (gfx1250) — matrix operand order repaired and checked by reading, not compiled, not run on hardware
//
#include <hip/hip_runtime.h>

// ---------- types ----------
typedef __attribute__((ext_vector_type(4)))  __bf16 v4bf;
typedef __attribute__((ext_vector_type(8)))  __bf16 v8bf;
typedef __attribute__((ext_vector_type(16))) __bf16 v16bf;
typedef __attribute__((ext_vector_type(8)))  float  v8f;

// ---------- helpers ----------
__device__ __forceinline__ __bf16 to_bf16(float f) {
  // round-to-nearest-even to bf16 via integer math (inputs finite)
  unsigned u = __builtin_bit_cast(unsigned, f);
  u += 0x7FFFu + ((u >> 16) & 1u);
  unsigned short h = (unsigned short)(u >> 16);
  return __builtin_bit_cast(__bf16, h);
}

// branch-free tanh: copysign(1 - 2/(e^{2|u|}+1), u); v_exp_f32 + v_rcp_f32
__device__ __forceinline__ float fast_tanh(float u) {
  float au = fabsf(u);
  float e  = __expf(2.0f * au);
  float r  = 1.0f - 2.0f * __builtin_amdgcn_rcpf(e + 1.0f);
  return copysignf(r, u);
}

__device__ __forceinline__ float gelu_tanh(float v) {
  // jax.nn.gelu default (approximate=True)
  const float k0 = 0.7978845608028654f;  // sqrt(2/pi)
  const float k1 = 0.044715f;
  float u = k0 * (v + k1 * v * v * v);
  return 0.5f * v * (1.0f + fast_tanh(u));
}

// CDNA5 async global->LDS copy (16B per lane), tracked by ASYNCcnt.
// Low 32 bits of a generic __shared__ pointer are the LDS byte offset
// (ISA 10.2: LDS aperture passes addr[31:0] through).
__device__ __forceinline__ void async_ld_b128(const __bf16* g, __bf16* l) {
  unsigned loff = (unsigned)(unsigned long long)l;
  asm volatile("global_load_async_to_lds_b128 %0, %1, off"
               :: "v"(loff), "v"(g) : "memory");
}
__device__ __forceinline__ void wait_async0() {
  asm volatile("s_wait_asynccnt 0x0" ::: "memory");
}

// ---------- conversion kernels ----------
// A[8192,2048] <- bf16(x[b, t, :]) for b in 0..3, t in 0..2047 (exact router
// result: softmax over a size-1 axis == 1.0, stable top_k -> first 2048 tokens)
__global__ __launch_bounds__(256) void gather_cvt_x(const float* __restrict__ x,
                                                    __bf16* __restrict__ A) {
  long i = ((long)blockIdx.x * blockDim.x + threadIdx.x) * 4;  // elem index in A
  int  m = (int)(i >> 11);      // row in A (K = 2048 = 2^11)
  int  k = (int)(i & 2047);
  long src = ((long)((m >> 11) * 4096 + (m & 2047))) * 2048 + k;  // b*S + t
  float4 v = *(const float4*)(x + src);
  v4bf o;
  o[0] = to_bf16(v.x); o[1] = to_bf16(v.y); o[2] = to_bf16(v.z); o[3] = to_bf16(v.w);
  *(v4bf*)(A + i) = o;
}

// D[C][R] <- bf16(S[R][C])  — tiled transpose so the GEMM's B operand is
// K-contiguous ([n][k]) and can be async-copied to LDS with no shuffles.
__global__ __launch_bounds__(256) void cvt_transpose_f32_bf16(
    const float* __restrict__ S, __bf16* __restrict__ D, int R, int C) {
  __shared__ float tile[32][33];
  const int tx = threadIdx.x & 31;
  const int ty = threadIdx.x >> 5;  // 0..7
  const long c0 = (long)blockIdx.x * 32;
  const long r0 = (long)blockIdx.y * 32;
#pragma unroll
  for (int i = 0; i < 4; ++i)
    tile[ty + i * 8][tx] = S[(r0 + ty + i * 8) * (long)C + c0 + tx];
  __syncthreads();
#pragma unroll
  for (int i = 0; i < 4; ++i)
    D[(c0 + ty + i * 8) * (long)R + r0 + tx] = to_bf16(tile[tx][ty + i * 8]);
}

// ---------- tiled bf16 WMMA GEMM ----------
// C[M,N] = epilogue(A[M,K] @ BT[N,K]^T + bias[N])
// EPI==0: gelu -> bf16 out ; EPI==1: identity -> f32 out
// 256 threads = 8 waves (2M x 4N), wave tile 64x64 = 4x4 WMMA fragments.
#define BM 128
#define BN 256
#define BK 32
#define LDST 40  // padded LDS stride in bf16 elements (80B -> conflict-free)

template <int EPI>
__global__ __launch_bounds__(256) void gemm_bf16_wmma(
    const __bf16* __restrict__ A,   // [M][K] row-major
    const __bf16* __restrict__ BT,  // [N][K] row-major (pre-transposed B)
    const float* __restrict__ bias, void* __restrict__ Cout,
    int M, int N, int K) {
  __shared__ __bf16 Al[2][BM][LDST];   // 20 KB
  __shared__ __bf16 Bl[2][BN][LDST];   // 40 KB

  const int tid   = threadIdx.x;
  const int lane  = tid & 31;  // wave32
  const int wid   = tid >> 5;  // 8 waves
  const int waveM = wid >> 2;  // 0..1 -> 64-row slice
  const int waveN = wid & 3;   // 0..3 -> 64-col slice
  const int half  = lane >> 4;
  const int r16   = lane & 15;

  const long bm = (long)blockIdx.y * BM;
  const long bn = (long)blockIdx.x * BN;

  // async staging: chunk = 16B (8 bf16). A tile: 512 chunks, B tile: 1024.
  // thread t moves A chunks {t, t+256} and B chunks {t, t+256, t+512, t+768}:
  // entity e = c>>2 (row/col), k-offset = (c&3)*8; +256 chunks => +64 entities.
  const int e0 = tid >> 2;
  const int ko = (tid & 3) * 8;

  const __bf16* agp = A  + (bm + e0) * (long)K + ko;
  const __bf16* bgp = BT + (bn + e0) * (long)K + ko;

  auto stage = [&](int kt, int buf) {
    long k0 = (long)kt * BK;
    async_ld_b128(agp + k0,                    &Al[buf][e0][ko]);
    async_ld_b128(agp + k0 + 64 * (long)K,     &Al[buf][e0 + 64][ko]);
#pragma unroll
    for (int q = 0; q < 4; ++q)
      async_ld_b128(bgp + k0 + (64L * q) * K,  &Bl[buf][e0 + 64 * q][ko]);
  };

  // A fragment (16x32): lanes0-15 row=r16, K {0..7,16..23}; lanes16-31 K {8..15,24..31}
  auto loadA = [&](int buf, int mf) -> v16bf {
    const __bf16* p = &Al[buf][mf * 16 + r16][half * 8];
    union { v16bf v; v8bf h[2]; } u;
    u.h[0] = *(const v8bf*)p;
    u.h[1] = *(const v8bf*)(p + 16);
    return u.v;
  };
  // B fragment (32x16): lanes0-15 col=r16 K {0..7,16..23}; lanes16-31 K {8..15,24..31}
  auto loadB = [&](int buf, int nf) -> v16bf {
    const __bf16* p = &Bl[buf][nf * 16 + r16][half * 8];
    union { v16bf v; v8bf h[2]; } u;
    u.h[0] = *(const v8bf*)p;
    u.h[1] = *(const v8bf*)(p + 16);
    return u.v;
  };

  v8f zero;
#pragma unroll
  for (int e = 0; e < 8; ++e) zero[e] = 0.0f;
  v8f acc[4][4];
#pragma unroll
  for (int i = 0; i < 4; ++i)
#pragma unroll
    for (int j = 0; j < 4; ++j) acc[i][j] = zero;

  const int nk = K / BK;
  stage(0, 0);
  wait_async0();
  __syncthreads();

  for (int kt = 0; kt < nk; ++kt) {
    const int cur = kt & 1;
    if (kt + 1 < nk) stage(kt + 1, cur ^ 1);   // async fill of next buffer
    if (kt + 4 < nk) {                         // L2 prefetch further ahead
      __builtin_prefetch(agp + (long)(kt + 4) * BK, 0, 0);
      __builtin_prefetch(bgp + (long)(kt + 4) * BK, 0, 0);
    }
    v16bf bfr[4];
#pragma unroll
    for (int j = 0; j < 4; ++j) bfr[j] = loadB(cur, waveN * 4 + j);
#pragma unroll
    for (int i = 0; i < 4; ++i) {
      v16bf af = loadA(cur, waveM * 4 + i);
#pragma unroll
      for (int j = 0; j < 4; ++j)
        acc[i][j] = __builtin_amdgcn_wmma_f32_16x16x32_bf16(
            false, af, false, bfr[j], (short)0, acc[i][j], false, false);
    }
    if (kt + 1 < nk) wait_async0();
    __syncthreads();
  }

  // epilogue: C layout — VGPR r: row = base + half*8 + r, col = base + r16
  float bcol[4];
#pragma unroll
  for (int j = 0; j < 4; ++j)
    bcol[j] = bias[bn + waveN * 64 + j * 16 + r16];

#pragma unroll
  for (int i = 0; i < 4; ++i) {
#pragma unroll
    for (int j = 0; j < 4; ++j) {
      long m0 = bm + waveM * 64 + i * 16 + half * 8;
      long n0 = bn + waveN * 64 + j * 16 + r16;
      v8f c = acc[i][j];
#pragma unroll
      for (int rr = 0; rr < 8; ++rr) {
        float v = c[rr] + bcol[j];
        if (EPI == 0) {
          ((__bf16*)Cout)[(m0 + rr) * (long)N + n0] = to_bf16(gelu_tanh(v));
        } else {
          ((float*)Cout)[(m0 + rr) * (long)N + n0] = v;
        }
      }
    }
  }
}

// ---------- launch ----------
extern "C" void kernel_launch(void* const* d_in, const int* in_sizes, int n_in,
                              void* d_out, int out_size, void* d_ws, size_t ws_size,
                              hipStream_t stream) {
  (void)in_sizes; (void)n_in; (void)out_size; (void)ws_size;
  // setup_inputs order: x, Wp, bp, W1, b1, W2, b2
  const float* x  = (const float*)d_in[0];
  const float* W1 = (const float*)d_in[3];
  const float* b1 = (const float*)d_in[4];
  const float* W2 = (const float*)d_in[5];
  const float* b2 = (const float*)d_in[6];
  float* out = (float*)d_out;

  const int M = 8192;  // 4 batches * 2048 selected tokens
  const int D = 2048;
  const int F = 8192;

  const size_t MB = 1024 * 1024;
  char* ws = (char*)d_ws;
  __bf16* Abf   = (__bf16*)(ws);            // 32 MB  [M][D]
  __bf16* W1tbf = (__bf16*)(ws + 32 * MB);  // 32 MB  [F][D]  (W1^T)
  __bf16* W2tbf = (__bf16*)(ws + 64 * MB);  // 32 MB  [D][F]  (W2^T)
  __bf16* Hbf   = (__bf16*)(ws + 96 * MB);  // 128 MB [M][F]

  // router is exact identity onto first 2048 tokens/batch: gather+convert
  gather_cvt_x<<<(unsigned)(((long)M * D) / 4 / 256), 256, 0, stream>>>(x, Abf);
  // pre-transpose weights to [n][k] so GEMM staging is pure async b128 copies
  cvt_transpose_f32_bf16<<<dim3(F / 32, D / 32), 256, 0, stream>>>(W1, W1tbf, D, F);
  cvt_transpose_f32_bf16<<<dim3(D / 32, F / 32), 256, 0, stream>>>(W2, W2tbf, F, D);

  dim3 g1(F / BN, M / BM);
  gemm_bf16_wmma<0><<<g1, 256, 0, stream>>>(Abf, W1tbf, b1, (void*)Hbf, M, F, D);
  dim3 g2(D / BN, M / BM);
  gemm_bf16_wmma<1><<<g2, 256, 0, stream>>>(Hbf, W2tbf, b2, (void*)out, M, D, F);
}
